// PureCartesianSparseE3ConvSave_66881230733965
// MI455X (gfx1250) — matrix operand order repaired and checked by reading, not compiled, not run on hardware
//
#include <hip/hip_runtime.h>

typedef __attribute__((ext_vector_type(16))) _Float16 v16h;
typedef __attribute__((ext_vector_type(8)))  float    v8f;

#define C1    8
#define NB    8
#define EMBD  16
#define WNUM  384
#define TOTD  416
#define RMAXF 5.0f
#define WPB   8   // waves per block

// ---------------- zero output ----------------
__global__ void zero_f32(float* __restrict__ p, long n) {
  long i = (long)blockIdx.x * blockDim.x + threadIdx.x;
  if (i < n) p[i] = 0.0f;
}

// ---------------- pack weights into WMMA B-fragment layout (f16) ----------------
// B 32x16 f16 fragment (this code's contiguous k order): lane L element i -> K = i + 16*(L/16), N = L%16.
// Fragment order: f=0..3   : layer1 ntile f        (K padded 8 -> 32)
//                 f=4..11  : layer2 (t = (f-4)>>1, kstep s = (f-4)&1)
//                 f=12..59 : layer3 (t = (f-12)>>1, kstep s = (f-12)&1)
__global__ void pack_wfrag(const float* __restrict__ w1,
                           const float* __restrict__ w2,
                           const float* __restrict__ w3,
                           _Float16* __restrict__ frag) {
  int idx = blockIdx.x * blockDim.x + threadIdx.x;
  if (idx >= 60 * 512) return;
  int f    = idx >> 9;
  int r    = idx & 511;
  int lane = r >> 4;
  int i    = r & 15;
  int khi  = (lane >= 16) ? 16 : 0;
  int ncol = lane & 15;
  float val = 0.0f;
  if (f < 4) {
    int k = i + khi;                 // kstep 0 only, pad K>=8 with zero
    int n = f * 16 + ncol;
    if (k < NB) val = w1[k * 64 + n];
  } else if (f < 12) {
    int local = f - 4;
    int t = local >> 1, s = local & 1;
    int k = 32 * s + i + khi;
    int n = t * 16 + ncol;
    val = w2[k * 64 + n];
  } else {
    int local = f - 12;
    int t = local >> 1, s = local & 1;
    int k = 32 * s + i + khi;
    int n = t * 16 + ncol;
    val = w3[k * WNUM + n];
  }
  frag[idx] = (_Float16)val;
}

__device__ __forceinline__ float silu_f(float x) {
  return x / (1.0f + __expf(-x));
}

// ---------------- per-node scalar MLP ----------------
__global__ void node_mlp(const int* __restrict__ A,
                         const float* __restrict__ emb,
                         const float* __restrict__ w1, const float* __restrict__ b1,
                         const float* __restrict__ w2, const float* __restrict__ b2,
                         float* __restrict__ Ai, int N) {
  int n = blockIdx.x * blockDim.x + threadIdx.x;
  if (n >= N) return;
  int a = A[n];
  float x[EMBD];
#pragma unroll
  for (int i = 0; i < EMBD; i++) x[i] = emb[a * EMBD + i];
  float h[64];
  for (int j = 0; j < 64; j++) {
    float acc = b1[j];
#pragma unroll
    for (int i = 0; i < EMBD; i++) acc = fmaf(x[i], w1[i * 64 + j], acc);
    h[j] = silu_f(acc);
  }
#pragma unroll
  for (int c = 0; c < C1; c++) {
    float acc = b2[c];
    for (int j = 0; j < 64; j++) acc = fmaf(h[j], w2[j * C1 + c], acc);
    Ai[(long)n * C1 + c] = acc;
  }
}

// ---------------- fused edge kernel: geometry + basis + WMMA MLP + TP + scatter ----------------
__global__ __launch_bounds__(256) void edge_tp_scatter(
    const float* __restrict__ pos,
    const int*   __restrict__ batch,
    const int*   __restrict__ esrc,
    const int*   __restrict__ edst,
    const float* __restrict__ shifts,
    const float* __restrict__ cell,
    const float* __restrict__ fb1,
    const float* __restrict__ fb2,
    const float* __restrict__ fb3,
    const float* __restrict__ Ai,
    const _Float16* __restrict__ frag,
    float* __restrict__ out,
    int E, float scale)
{
  __shared__ _Float16 Hs[WPB][16][72];      // activation tile [edge m][hidden k]
  __shared__ float    AiS[WPB][16][C1];
  __shared__ float    uS[WPB][16][4];
  __shared__ int      dS[WPB][16];

  const int lane = threadIdx.x & 31;
  const int wv   = threadIdx.x >> 5;
  const int m    = lane & 15;      // edge row within tile
  const int hi   = lane >> 4;      // half-wave selector
  const int ncol = m;              // output column for D fragments

  const int numTiles = (E + 15) >> 4;
  int tile = blockIdx.x * WPB + wv;
  const bool tv = (tile < numTiles);
  if (!tv) tile = 0;
  int e = tile * 16 + m;
  if (e >= E) e = 0;

  const int src = esrc[e];
  const int dst = edst[e];
  const int bi  = batch[src];

  // ---- geometry (PBC shift @ cell) ----
  float u3[3], vlen;
  {
    float sh0 = shifts[e * 3 + 0], sh1 = shifts[e * 3 + 1], sh2 = shifts[e * 3 + 2];
    float v[3];
#pragma unroll
    for (int j = 0; j < 3; j++) {
      float sv = sh0 * cell[bi * 9 + 0 + j] + sh1 * cell[bi * 9 + 3 + j] + sh2 * cell[bi * 9 + 6 + j];
      v[j] = pos[dst * 3 + j] - pos[src * 3 + j] + sv;
    }
    vlen = sqrtf(v[0] * v[0] + v[1] * v[1] + v[2] * v[2] + 1e-12f);
    float inv = 1.0f / vlen;
    u3[0] = v[0] * inv; u3[1] = v[1] * inv; u3[2] = v[2] * inv;
  }

  // ---- radial basis: gaussian soft-one-hot * sqrt(NB)/1.12 ----
  const float step = RMAXF / (float)(NB + 1);
  const float coef = 2.52538136f;  // sqrt(8)/1.12
  float bas[NB];
#pragma unroll
  for (int k = 0; k < NB; k++) {
    float d = (vlen - step * (float)(k + 1)) / step;
    bas[k] = __expf(-d * d) * coef;
  }

  if (hi == 0) {
    dS[wv][m] = dst;
    uS[wv][m][0] = u3[0]; uS[wv][m][1] = u3[1]; uS[wv][m][2] = u3[2];
#pragma unroll
    for (int c = 0; c < C1; c++) AiS[wv][m][c] = Ai[(long)src * C1 + c];
  }

  // ---- A fragment for layer1 (16x32 f16, K padded: only K<8 nonzero) ----
  // ISA layout: lanes 0-15 elements 0..7 hold K=0..7; all other (lane,elem) map to K>=8 -> 0
  v16h a1 = {};
  if (hi == 0) {
#pragma unroll
    for (int i = 0; i < NB; i++) a1[i] = (_Float16)bas[i];
  }
  __syncthreads();

  // ---- layer 1: [16,32(pad)] x [32,64] ----
#pragma unroll
  for (int nt = 0; nt < 4; nt++) {
    v16h b = *(const v16h*)(frag + (size_t)nt * 512 + lane * 16);
    v8f c = {};
    c = __builtin_amdgcn_wmma_f32_16x16x32_f16(false, a1, false, b, (short)0, c, false, false);
    float bias = fb1[nt * 16 + ncol];
#pragma unroll
    for (int r = 0; r < 8; r++)
      Hs[wv][r + 8 * hi][nt * 16 + ncol] = (_Float16)silu_f(c[r] + bias);
  }
  __syncthreads();

  // ---- gather A fragments for layer 2 (K=64 -> 2 ksteps) ----
  // A layout: lane L element i -> K = i + 16*(L/16), row M = L%16
  v16h a2[2];
#pragma unroll
  for (int s = 0; s < 2; s++)
#pragma unroll
    for (int i = 0; i < 16; i++) {
      int k = i + 16 * hi + 32 * s;
      a2[s][i] = Hs[wv][m][k];
    }
  __syncthreads();

  // ---- layer 2: [16,64] x [64,64] (in-place H overwrite after full read) ----
#pragma unroll
  for (int nt = 0; nt < 4; nt++) {
    v8f c = {};
#pragma unroll
    for (int s = 0; s < 2; s++) {
      v16h b = *(const v16h*)(frag + (size_t)(4 + nt * 2 + s) * 512 + lane * 16);
      c = __builtin_amdgcn_wmma_f32_16x16x32_f16(false, a2[s], false, b, (short)0, c, false, false);
    }
    float bias = fb2[nt * 16 + ncol];
#pragma unroll
    for (int r = 0; r < 8; r++)
      Hs[wv][r + 8 * hi][nt * 16 + ncol] = (_Float16)silu_f(c[r] + bias);
  }
  __syncthreads();

  v16h a3[2];
#pragma unroll
  for (int s = 0; s < 2; s++)
#pragma unroll
    for (int i = 0; i < 16; i++) {
      int k = i + 16 * hi + 32 * s;
      a3[s][i] = Hs[wv][m][k];
    }

  // ---- layer 3 + channel contraction fused ----
  // Output tile t covers n in [16t,16t+16): l = t>>3, c = t&7 constant; o = lane%16.
  // s_l[row, o] += (w + b3) * Ai[row, c] accumulates entirely in-register per lane.
  float s0[8], s1[8], s2[8];
#pragma unroll
  for (int r = 0; r < 8; r++) { s0[r] = 0.f; s1[r] = 0.f; s2[r] = 0.f; }

  for (int t = 0; t < 24; t++) {
    v8f d = {};
#pragma unroll
    for (int s = 0; s < 2; s++) {
      v16h b = *(const v16h*)(frag + (size_t)(12 + t * 2 + s) * 512 + lane * 16);
      d = __builtin_amdgcn_wmma_f32_16x16x32_f16(false, a3[s], false, b, (short)0, d, false, false);
    }
    float bias = fb3[t * 16 + ncol];
    int l = t >> 3, cc = t & 7;
    float* acc = (l == 0) ? s0 : (l == 1) ? s1 : s2;
#pragma unroll
    for (int r = 0; r < 8; r++)
      acc[r] += (d[r] + bias) * AiS[wv][r + 8 * hi][cc];
  }

  // ---- scatter: rank-0/1/2 cartesian features, normalized, atomic segment-sum ----
#pragma unroll
  for (int r = 0; r < 8; r++) {
    int mr = r + 8 * hi;
    int eg = tile * 16 + mr;
    if (!tv || eg >= E) continue;
    int dd = dS[wv][mr];
    float ux = uS[wv][mr][0], uy = uS[wv][mr][1], uz = uS[wv][mr][2];
    float* po = out + (size_t)dd * TOTD;
    atomicAdd(po + ncol, s0[r] * scale);
    float w1v = s1[r] * scale;
    atomicAdd(po + 16 + ncol * 3 + 0, w1v * ux);
    atomicAdd(po + 16 + ncol * 3 + 1, w1v * uy);
    atomicAdd(po + 16 + ncol * 3 + 2, w1v * uz);
    float w2v = s2[r] * scale;
    float uu[3] = {ux, uy, uz};
#pragma unroll
    for (int i2 = 0; i2 < 3; i2++)
#pragma unroll
      for (int j2 = 0; j2 < 3; j2++)
        atomicAdd(po + 64 + ncol * 9 + 3 * i2 + j2, w2v * uu[i2] * uu[j2]);
  }
}

extern "C" void kernel_launch(void* const* d_in, const int* in_sizes, int n_in,
                              void* d_out, int out_size, void* d_ws, size_t ws_size,
                              hipStream_t stream) {
  const float* pos    = (const float*)d_in[0];
  const int*   A      = (const int*)  d_in[1];
  const int*   batch  = (const int*)  d_in[2];
  const int*   esrc   = (const int*)  d_in[3];
  const int*   edst   = (const int*)  d_in[4];
  const float* shifts = (const float*)d_in[5];
  const float* cell   = (const float*)d_in[6];
  const float* emb    = (const float*)d_in[7];
  const float* aw1    = (const float*)d_in[8];
  const float* ab1    = (const float*)d_in[9];
  const float* aw2    = (const float*)d_in[10];
  const float* ab2    = (const float*)d_in[11];
  const float* fw1    = (const float*)d_in[12];
  const float* fb1    = (const float*)d_in[13];
  const float* fw2    = (const float*)d_in[14];
  const float* fb2    = (const float*)d_in[15];
  const float* fw3    = (const float*)d_in[16];
  const float* fb3    = (const float*)d_in[17];
  float* out = (float*)d_out;
  const int N = in_sizes[1];
  const int E = in_sizes[3];

  // workspace: Ai [N,8] f32, then 60 B-fragments (512 f16 each)
  char* ws = (char*)d_ws;
  float* Ai = (float*)ws;
  size_t aiBytes = (((size_t)N * C1 * sizeof(float)) + 255) & ~(size_t)255;
  _Float16* frag = (_Float16*)(ws + aiBytes);

  long totOut = (long)N * TOTD;
  zero_f32<<<(unsigned)((totOut + 255) / 256), 256, 0, stream>>>(out, totOut);
  pack_wfrag<<<(60 * 512 + 255) / 256, 256, 0, stream>>>(fw1, fw2, fw3, frag);
  node_mlp<<<(N + 255) / 256, 256, 0, stream>>>(A, emb, aw1, ab1, aw2, ab2, Ai, N);

  float avg = (float)E / (float)N;
  if (avg < 1e-8f) avg = 1e-8f;
  float scale = 1.0f / avg;
  int numTiles = (E + 15) >> 4;
  int blocks = (numTiles + WPB - 1) / WPB;
  edge_tp_scatter<<<blocks, 256, 0, stream>>>(pos, batch, esrc, edst, shifts, cell,
                                              fb1, fb2, fb3, Ai, frag, out, E, scale);
}
